// SelfAttention_68702296867634
// MI455X (gfx1250) — compile-verified, matrix operand order checked by reading, not run on hardware
//
#include <hip/hip_runtime.h>

#define NSTATE 384
#define NHEAD  6
#define HD     64
#define BATCH  128
#define SEQ    448
#define NROWS  (BATCH * SEQ)   // 57344

typedef __attribute__((ext_vector_type(16))) _Float16 v16h;
typedef __attribute__((ext_vector_type(8)))  _Float16 v8h;
typedef __attribute__((ext_vector_type(8)))  float    v8f;

__device__ __forceinline__ v8f vzero() {
  v8f v;
#pragma unroll
  for (int i = 0; i < 8; ++i) v[i] = 0.0f;
  return v;
}

// Load a 16x32 f16 A/B fragment. p = row base + kbase + (lane>>4)*8.
// Elements 0..7  <- K = kbase + half*8 .. +7      (VGPRs 0-3 of ISA layout)
// Elements 8..15 <- K = kbase + 16 + half*8 .. +7 (VGPRs 4-7 of ISA layout)
__device__ __forceinline__ v16h frag_ld(const _Float16* p) {
  v8h lo = *(const v8h*)(p);
  v8h hi = *(const v8h*)(p + 16);
  v16h f;
#pragma unroll
  for (int i = 0; i < 8; ++i) { f[i] = lo[i]; f[i + 8] = hi[i]; }
  return f;
}

// Same fragment, but from an f32 LDS tile with on-the-fly f32->f16 convert.
__device__ __forceinline__ v16h frag_ld_f32(const float* p) {
  const float4 a0 = *(const float4*)(p + 0);
  const float4 a1 = *(const float4*)(p + 4);
  const float4 b0 = *(const float4*)(p + 16);
  const float4 b1 = *(const float4*)(p + 20);
  v16h f;
  f[0]  = (_Float16)a0.x; f[1]  = (_Float16)a0.y;
  f[2]  = (_Float16)a0.z; f[3]  = (_Float16)a0.w;
  f[4]  = (_Float16)a1.x; f[5]  = (_Float16)a1.y;
  f[6]  = (_Float16)a1.z; f[7]  = (_Float16)a1.w;
  f[8]  = (_Float16)b0.x; f[9]  = (_Float16)b0.y;
  f[10] = (_Float16)b0.z; f[11] = (_Float16)b0.w;
  f[12] = (_Float16)b1.x; f[13] = (_Float16)b1.y;
  f[14] = (_Float16)b1.z; f[15] = (_Float16)b1.w;
  return f;
}

__device__ __forceinline__ v8f wmma16(v16h a, v16h b, v8f c) {
  return __builtin_amdgcn_wmma_f32_16x16x32_f16(false, a, false, b, (short)0, c,
                                                false, false);
}

// Async global -> LDS copies (CDNA5, tracked with ASYNCcnt).
// lds_off = low 32 bits of the generic LDS address (== LDS byte offset).
__device__ __forceinline__ void async_b128(unsigned lds_off, const void* g) {
  asm volatile("global_load_async_to_lds_b128 %0, %1, off"
               :: "v"(lds_off), "v"(g) : "memory");
}
__device__ __forceinline__ void async_b32(unsigned lds_off, const void* g) {
  asm volatile("global_load_async_to_lds_b32 %0, %1, off"
               :: "v"(lds_off), "v"(g) : "memory");
}
__device__ __forceinline__ void wait_async_le10() {
#if defined(__has_builtin) && __has_builtin(__builtin_amdgcn_s_wait_asynccnt)
  __builtin_amdgcn_s_wait_asynccnt(10);
#else
  asm volatile("s_wait_asynccnt 0xa" ::: "memory");
#endif
}

// ---------------------------------------------------------------------------
// Kernel 1: convert Wq/Wk/Wv/Wo (f32 [N][K] row-major) to f16 in workspace.
// ---------------------------------------------------------------------------
__global__ void cvt_weights(const float* __restrict__ Wq, const float* __restrict__ Wk,
                            const float* __restrict__ Wv, const float* __restrict__ Wo,
                            _Float16* __restrict__ out) {
  int i = blockIdx.x * 256 + threadIdx.x;
  const int M = NSTATE * NSTATE;
  if (i >= 4 * M) return;
  const float* src = (i < M) ? Wq : (i < 2 * M) ? Wk : (i < 3 * M) ? Wv : Wo;
  out[i] = (_Float16)src[i % M];
}

// ---------------------------------------------------------------------------
// Kernel 2: h = emb[x] (gather -> f16 LDS), then q/k/v = h @ W^T (+bias).
// q (scaled by 1/8 = d^-0.5 folded) -> ws f16; k,v (f32) -> d_out.
// ---------------------------------------------------------------------------
__global__ __launch_bounds__(256) void qkv_kernel(
    const int* __restrict__ x, const float* __restrict__ emb,
    const _Float16* __restrict__ W16, const float* __restrict__ bq,
    const float* __restrict__ bv, _Float16* __restrict__ q16,
    float* __restrict__ outK, float* __restrict__ outV) {
  __shared__ __align__(16) _Float16 hbuf[64 * 392];  // 196 dw stride % 64 = 4
  const int t = threadIdx.x;
  const int rowbase = blockIdx.x * 64;

  for (int i = t; i < 64 * NSTATE; i += 256) {
    int r = i / NSTATE, c = i - r * NSTATE;
    int tok = x[rowbase + r];
    hbuf[r * 392 + c] = (_Float16)emb[(size_t)tok * NSTATE + c];
  }
  __syncthreads();

  const int wave = t >> 5, lane = t & 31;
  const int lr = lane & 15, half = lane >> 4;
  const int mt = wave & 3;
  const int ng0 = (wave >> 2) * 3;

  for (int mat = 0; mat < 3; ++mat) {
    const _Float16* W = W16 + (size_t)mat * NSTATE * NSTATE;
    for (int ng = ng0; ng < ng0 + 3; ++ng) {
      v8f acc[4];
#pragma unroll
      for (int nt = 0; nt < 4; ++nt) acc[nt] = vzero();

      for (int ks = 0; ks < 12; ++ks) {
        v16h a = frag_ld(&hbuf[(mt * 16 + lr) * 392 + ks * 32 + half * 8]);
#pragma unroll
        for (int nt = 0; nt < 4; ++nt) {
          int n = ng * 64 + nt * 16 + lr;
          v16h b = frag_ld(W + (size_t)n * NSTATE + ks * 32 + half * 8);
          acc[nt] = wmma16(a, b, acc[nt]);
        }
      }
#pragma unroll
      for (int nt = 0; nt < 4; ++nt) {
        int col = ng * 64 + nt * 16 + lr;         // C layout: N = lane&15
#pragma unroll
        for (int j = 0; j < 8; ++j) {             // C layout: M = j + 8*half
          int row = rowbase + mt * 16 + j + 8 * half;
          size_t off = (size_t)row * NSTATE + col;
          float v = acc[nt][j];
          if (mat == 0)      q16[off] = (_Float16)((v + bq[col]) * 0.125f);
          else if (mat == 1) outK[off] = v;
          else               outV[off] = v + bv[col];
        }
      }
    }
  }
}

// ---------------------------------------------------------------------------
// Kernel 3: flash attention with double-buffered async global->LDS K/V tiles.
// Block = (b, h, 128-query chunk); wave owns one 16-row q-tile.
// ---------------------------------------------------------------------------
__global__ __launch_bounds__(256) void attn_kernel(
    const _Float16* __restrict__ q16, const float* __restrict__ Kg,
    const float* __restrict__ Vg, _Float16* __restrict__ wv16) {
  __shared__ __align__(16) float Kf[2][32 * 72];      // [key][d] f32, stride 72
  __shared__ __align__(16) float Vf[2][64 * 40];      // [d][key] f32, stride 40
  __shared__ __align__(16) _Float16 Ps[8 * 16 * 40];  // per-wave P staging

  const int t = threadIdx.x, wave = t >> 5, lane = t & 31;
  const int lr = lane & 15, half = lane >> 4;
  const int bid = blockIdx.x;
  const int qc = bid & 3, bh = bid >> 2;
  const int h = bh % NHEAD, b = bh / NHEAD;
  const int qt = qc * 8 + wave;        // q-tile 0..31 (28 valid for T=448)
  const bool valid = qt < 28;
  const int qtc = valid ? qt : 27;

  // q A-fragments (16x64 -> two 16x32 frags), kept in registers
  const size_t qrow = (size_t)(b * SEQ + qtc * 16 + lr) * NSTATE + h * HD;
  v16h qa0 = frag_ld(q16 + qrow + 0 + half * 8);
  v16h qa1 = frag_ld(q16 + qrow + 32 + half * 8);

  float m[8], lsum[8];
  v8f acc[4];
#pragma unroll
  for (int j = 0; j < 8; ++j) { m[j] = -1e30f; lsum[j] = 0.0f; }
#pragma unroll
  for (int nt = 0; nt < 4; ++nt) acc[nt] = vzero();

  int qend = (qc + 1) * 128;
  if (qend > SEQ) qend = SEQ;
  const int nkt = qend / 32;

  // Issue async copies of one 32-key K/V tile (10 async ops per thread).
  auto issue_tile = [&](int ktile, int bsel) {
    const float* kb = Kg + (size_t)(b * SEQ + ktile * 32) * NSTATE + h * HD;
    const float* vb = Vg + (size_t)(b * SEQ + ktile * 32) * NSTATE + h * HD;
#pragma unroll
    for (int i = 0; i < 2; ++i) {            // K: 512 16B chunks, row-major
      int idx = t + i * 256;
      int key = idx >> 4, c = idx & 15;
      unsigned l = (unsigned)(uintptr_t)&Kf[bsel][key * 72 + c * 4];
      async_b128(l, kb + (size_t)key * NSTATE + c * 4);
    }
#pragma unroll
    for (int i = 0; i < 8; ++i) {            // V: 2048 scattered b32 (transpose)
      int idx = t + i * 256;
      int key = idx >> 6, d = idx & 63;
      unsigned l = (unsigned)(uintptr_t)&Vf[bsel][d * 40 + key];
      async_b32(l, vb + (size_t)key * NSTATE + d);
    }
  };

  issue_tile(0, 0);

  for (int kt = 0; kt < nkt; ++kt) {
    // Prefetch next tile into the other buffer (clamped redundant issue on
    // the last step keeps the ASYNCcnt bookkeeping uniform).
    int nx = (kt + 1 < nkt) ? kt + 1 : nkt - 1;
    issue_tile(nx, (kt + 1) & 1);
    wait_async_le10();                 // own 10 ops of tile kt complete
    __syncthreads();                   // -> whole tile visible to all waves

    const float* Kb = Kf[kt & 1];
    const float* Vb = Vf[kt & 1];

    // S = q . K^T  (d^-0.5 already folded into q16)
    v8f s[2];
#pragma unroll
    for (int nf = 0; nf < 2; ++nf) {
      const float* kp = &Kb[(nf * 16 + lr) * 72 + half * 8];
      v8f c = vzero();
      c = wmma16(qa0, frag_ld_f32(kp), c);
      c = wmma16(qa1, frag_ld_f32(kp + 32), c);
      s[nf] = c;
    }
    // causal mask using C layout (M = j + 8*half, N = lr)
    const int q0 = qtc * 16 + 8 * half;
#pragma unroll
    for (int nf = 0; nf < 2; ++nf) {
      int key = kt * 32 + nf * 16 + lr;
#pragma unroll
      for (int j = 0; j < 8; ++j)
        if (key > q0 + j) s[nf][j] = -1e30f;
    }
    // online softmax: row max over the 16 lanes of each half-wave
    float rmax[8];
#pragma unroll
    for (int j = 0; j < 8; ++j) {
      rmax[j] = fmaxf(s[0][j], s[1][j]);
      rmax[j] = fmaxf(rmax[j], __shfl_xor(rmax[j], 1, 32));
      rmax[j] = fmaxf(rmax[j], __shfl_xor(rmax[j], 2, 32));
      rmax[j] = fmaxf(rmax[j], __shfl_xor(rmax[j], 4, 32));
      rmax[j] = fmaxf(rmax[j], __shfl_xor(rmax[j], 8, 32));
    }
    float corr[8];
#pragma unroll
    for (int j = 0; j < 8; ++j) {
      float mn = fmaxf(m[j], rmax[j]);
      corr[j] = __expf(m[j] - mn);
      m[j] = mn;
    }
    // P = exp(S - m); stage f16 to convert C-layout -> A-layout
    _Float16* stg = &Ps[wave * 640];
#pragma unroll
    for (int nf = 0; nf < 2; ++nf) {
#pragma unroll
      for (int j = 0; j < 8; ++j) {
        float p = __expf(s[nf][j] - m[j]);
        s[nf][j] = p;
        stg[(j + 8 * half) * 40 + nf * 16 + lr] = (_Float16)p;
      }
    }
#pragma unroll
    for (int j = 0; j < 8; ++j) {
      float rs = s[0][j] + s[1][j];
      rs += __shfl_xor(rs, 1, 32);
      rs += __shfl_xor(rs, 2, 32);
      rs += __shfl_xor(rs, 4, 32);
      rs += __shfl_xor(rs, 8, 32);
      lsum[j] = lsum[j] * corr[j] + rs;
    }
#pragma unroll
    for (int nt = 0; nt < 4; ++nt)
#pragma unroll
      for (int j = 0; j < 8; ++j) acc[nt][j] *= corr[j];

    // O += P x V  (P as A-frag 16x32; V^T rows give B columns)
    v16h pa = frag_ld(&stg[lr * 40 + half * 8]);
#pragma unroll
    for (int nt = 0; nt < 4; ++nt) {
      v16h vb2 = frag_ld_f32(&Vb[(nt * 16 + lr) * 40 + half * 8]);
      acc[nt] = wmma16(pa, vb2, acc[nt]);
    }
    __syncthreads();   // all waves done with tile kt before its buffer reuse
  }

  if (valid) {
#pragma unroll
    for (int nt = 0; nt < 4; ++nt) {
      int col = h * HD + nt * 16 + lr;
#pragma unroll
      for (int j = 0; j < 8; ++j) {
        int row = b * SEQ + qt * 16 + j + 8 * half;
        wv16[(size_t)row * NSTATE + col] = (_Float16)(acc[nt][j] / lsum[j]);
      }
    }
  }
}

// ---------------------------------------------------------------------------
// Kernel 4: out = wv @ Wo^T + bo  (f16 GEMM, f32 out)
// ---------------------------------------------------------------------------
__global__ __launch_bounds__(256) void oproj_kernel(
    const _Float16* __restrict__ wv16, const _Float16* __restrict__ Wo16,
    const float* __restrict__ bo, float* __restrict__ outO) {
  __shared__ __align__(16) _Float16 abuf[64 * 392];
  const int t = threadIdx.x;
  const int rowbase = blockIdx.x * 64;

  const uint32_t* src = (const uint32_t*)(wv16 + (size_t)rowbase * NSTATE);
  uint32_t* dst = (uint32_t*)abuf;
  for (int i = t; i < 64 * 192; i += 256) {
    int r = i / 192, c = i - r * 192;
    dst[r * 196 + c] = src[i];
  }
  __syncthreads();

  const int wave = t >> 5, lane = t & 31;
  const int lr = lane & 15, half = lane >> 4;
  const int mt = wave & 3;
  const int ng0 = (wave >> 2) * 3;

  for (int ng = ng0; ng < ng0 + 3; ++ng) {
    v8f acc[4];
#pragma unroll
    for (int nt = 0; nt < 4; ++nt) acc[nt] = vzero();
    for (int ks = 0; ks < 12; ++ks) {
      v16h a = frag_ld(&abuf[(mt * 16 + lr) * 392 + ks * 32 + half * 8]);
#pragma unroll
      for (int nt = 0; nt < 4; ++nt) {
        int n = ng * 64 + nt * 16 + lr;
        v16h b = frag_ld(Wo16 + (size_t)n * NSTATE + ks * 32 + half * 8);
        acc[nt] = wmma16(a, b, acc[nt]);
      }
    }
#pragma unroll
    for (int nt = 0; nt < 4; ++nt) {
      int col = ng * 64 + nt * 16 + lr;
#pragma unroll
      for (int j = 0; j < 8; ++j) {
        int row = rowbase + mt * 16 + j + 8 * half;
        outO[(size_t)row * NSTATE + col] = acc[nt][j] + bo[col];
      }
    }
  }
}

// ---------------------------------------------------------------------------
extern "C" void kernel_launch(void* const* d_in, const int* in_sizes, int n_in,
                              void* d_out, int out_size, void* d_ws, size_t ws_size,
                              hipStream_t stream) {
  (void)in_sizes; (void)n_in; (void)out_size; (void)ws_size;
  const int*   x   = (const int*)d_in[0];
  const float* emb = (const float*)d_in[1];
  const float* Wq  = (const float*)d_in[2];
  const float* bq  = (const float*)d_in[3];
  const float* Wk  = (const float*)d_in[4];
  const float* Wv  = (const float*)d_in[5];
  const float* bv  = (const float*)d_in[6];
  const float* Wo  = (const float*)d_in[7];
  const float* bo  = (const float*)d_in[8];

  float* outO = (float*)d_out;
  float* outK = outO + (size_t)NROWS * NSTATE;
  float* outV = outK + (size_t)NROWS * NSTATE;

  _Float16* wsW  = (_Float16*)d_ws;                       // 4 * 384*384 f16
  _Float16* q16  = wsW + (size_t)4 * NSTATE * NSTATE;     // [NROWS][384] f16
  _Float16* wv16 = q16 + (size_t)NROWS * NSTATE;          // [NROWS][384] f16
  _Float16* Wo16 = wsW + (size_t)3 * NSTATE * NSTATE;

  cvt_weights<<<(4 * NSTATE * NSTATE + 255) / 256, 256, 0, stream>>>(Wq, Wk, Wv, Wo, wsW);
  qkv_kernel<<<NROWS / 64, 256, 0, stream>>>(x, emb, wsW, bq, bv, q16, outK, outV);
  attn_kernel<<<BATCH * NHEAD * 4, 256, 0, stream>>>(q16, outK, outV, wv16);
  oproj_kernel<<<NROWS / 64, 256, 0, stream>>>(wv16, Wo16, bo, outO);
}
